// SparseIndexQPLayer_74955769249992
// MI455X (gfx1250) — matrix operand order repaired and checked by reading, not compiled, not run on hardware
//
#include <hip/hip_runtime.h>

// ---------------------------------------------------------------------------
// SparseIndexQPLayer for MI455X (gfx1250, wave32, WMMA)
//
// Phase 1: Q_b = R_bᵀ R_b  (256 x [500x252]·[252x500]) via v_wmma_f32_16x16x32_bf16.
//          Q stored as bf16, row-stride 512  => 256*512*512*2B = 134 MB in d_ws.
//          bf16 storage makes all 256 Q matrices resident in the 192 MB L2,
//          so the 330 subsequent mat-vec sweeps never touch HBM.
// Phase 2: per-batch persistent workgroup: 30 power iterations (Lipschitz L),
//          300 FISTA steps with simplex projection by bisection on theta.
// ---------------------------------------------------------------------------

#define BATCH 256
#define LOOK  252
#define NA    500
#define NP    512   // padded leading dimension of Q (bf16 elements)
#define SLDK  40    // LDS tile row stride in shorts (80 B: 16B-aligned rows,
                    // uint bank stride 20 -> near conflict-free)

typedef __attribute__((ext_vector_type(16))) __bf16 v16bf;
typedef __attribute__((ext_vector_type(8)))  float  v8f;

union Frag16 {
    unsigned int u[8];
    v16bf        v;
};

__device__ __forceinline__ unsigned short f2bf(float f) {
    unsigned int u = __float_as_uint(f);
    u += 0x7FFFu + ((u >> 16) & 1u);          // round-to-nearest-even
    return (unsigned short)(u >> 16);
}

// ---------------------------------------------------------------------------
// Kernel 1: Q = R^T R  (per batch), bf16 WMMA, 64x64 C tile per workgroup.
// 8 waves: wave w computes rows m_off=(w&3)*16, cols n_off=(w>>2)*32 (2 tiles).
// ---------------------------------------------------------------------------
__global__ __launch_bounds__(256)
void qp_build_Q(const float* __restrict__ R, unsigned short* __restrict__ Q)
{
    __shared__ unsigned short sA[64 * SLDK];   // [m][k] bf16, zero-padded
    __shared__ unsigned short sB[64 * SLDK];   // [n][k] bf16, zero-padded

    const int b    = blockIdx.z;
    const int i0   = blockIdx.y * 64;        // C row tile
    const int j0   = blockIdx.x * 64;        // C col tile
    const int tid  = threadIdx.x;
    const int lane = tid & 31;
    const int wid  = tid >> 5;
    const int m_off = (wid & 3) * 16;
    const int n_off = (wid >> 2) * 32;

    v8f c0, c1;
#pragma unroll
    for (int i = 0; i < 8; ++i) { c0[i] = 0.0f; c1[i] = 0.0f; }

    const size_t rbase = (size_t)b * LOOK * NA;

    for (int k0 = 0; k0 < LOOK; k0 += 32) {
        // ---- stage A tile: A[m][k] = R[k0+k][i0+m]  (branchless, clamped) ----
#pragma unroll
        for (int t = 0; t < 8; ++t) {
            const int idx = tid + t * 256;
            const int m = idx & 63, k = idx >> 6;
            const int gi = i0 + m, gk = k0 + k;
            const int giC = min(gi, NA - 1);
            const int gkC = min(gk, LOOK - 1);
            float fv = R[rbase + (size_t)gkC * NA + giC];   // always in-bounds
            fv = (gi < NA && gk < LOOK) ? fv : 0.0f;        // v_cndmask, no branch
            sA[m * SLDK + k] = f2bf(fv);
        }
        // ---- stage B tile: B[n][k] = R[k0+k][j0+n] ----
#pragma unroll
        for (int t = 0; t < 8; ++t) {
            const int idx = tid + t * 256;
            const int n = idx & 63, k = idx >> 6;
            const int gj = j0 + n, gk = k0 + k;
            const int gjC = min(gj, NA - 1);
            const int gkC = min(gk, LOOK - 1);
            float fv = R[rbase + (size_t)gkC * NA + gjC];
            fv = (gj < NA && gk < LOOK) ? fv : 0.0f;
            sB[n * SLDK + k] = f2bf(fv);
        }
        __syncthreads();

        const unsigned int* pA = (const unsigned int*)sA;
        const unsigned int* pB = (const unsigned int*)sB;
        const int khalf = lane >> 4;

        // A fragment, 16-bit A 16x32 layout:
        // lanes 0-15: K 0-7 (v0-3), 16-23 (v4-7); lanes 16-31: +8.
        Frag16 fa;
        {
            const int mrow = m_off + (lane & 15);
#pragma unroll
            for (int p = 0; p < 8; ++p) {
                const int k = ((p < 4) ? 0 : 16) + khalf * 8 + (p & 3) * 2;
                fa.u[p] = pA[mrow * (SLDK / 2) + (k >> 1)];
            }
        }
        // B fragments, same K order as A: lanes 0-15: K 0-7, 16-23; lanes 16-31: +8. K packed 2/VGPR.
        Frag16 fb0, fb1;
        {
            const int ncol = n_off + (lane & 15);
#pragma unroll
            for (int p = 0; p < 8; ++p) {
                const int k = ((p < 4) ? 0 : 16) + khalf * 8 + (p & 3) * 2;
                fb0.u[p] = pB[ncol * (SLDK / 2) + (k >> 1)];
                fb1.u[p] = pB[(ncol + 16) * (SLDK / 2) + (k >> 1)];
            }
        }

        c0 = __builtin_amdgcn_wmma_f32_16x16x32_bf16(false, fa.v, false, fb0.v,
                                                     (short)0, c0, false, false);
        c1 = __builtin_amdgcn_wmma_f32_16x16x32_bf16(false, fa.v, false, fb1.v,
                                                     (short)0, c1, false, false);
        __syncthreads();
    }

    // ---- store C: VGPR v -> row v + 8*(lane>=16), col = lane&15 ----
    const size_t qbase = (size_t)b * NP * NP;
#pragma unroll
    for (int v = 0; v < 8; ++v) {
        const int i = i0 + m_off + v + (lane >> 4) * 8;
        const int j = j0 + n_off + (lane & 15);
        if (i < NA) {
            if (j < NA)      Q[qbase + (size_t)i * NP + j]      = f2bf(c0[v]);
            if (j + 16 < NA) Q[qbase + (size_t)i * NP + j + 16] = f2bf(c1[v]);
        }
    }
}

// ---------------------------------------------------------------------------
// Kernel 2: per-batch FISTA solver. 512 threads = 16 waves; thread t owns
// element t of all length-500 vectors. Q rows read as packed bf16 quads.
// ---------------------------------------------------------------------------
__device__ __forceinline__ float blockSum(float v, float* red, int tid)
{
#pragma unroll
    for (int m = 16; m >= 1; m >>= 1) v += __shfl_xor(v, m, 32);
    const int wid = tid >> 5, lane = tid & 31;
    if (lane == 0) red[wid] = v;
    __syncthreads();
    if (tid == 0) {
        float s = 0.0f;
        for (int i = 0; i < 16; ++i) s += red[i];
        red[16] = s;
    }
    __syncthreads();
    return red[16];
}

__device__ __forceinline__ float blockMax(float v, float* red, int tid)
{
#pragma unroll
    for (int m = 16; m >= 1; m >>= 1) v = fmaxf(v, __shfl_xor(v, m, 32));
    const int wid = tid >> 5, lane = tid & 31;
    if (lane == 0) red[wid] = v;
    __syncthreads();
    if (tid == 0) {
        float s = red[0];
        for (int i = 1; i < 16; ++i) s = fmaxf(s, red[i]);
        red[16] = s;
    }
    __syncthreads();
    return red[16];
}

__device__ __forceinline__ float dot4_bf16(uint2 q, const float* __restrict__ x, int j)
{
    return __uint_as_float(q.x << 16)         * x[j]
         + __uint_as_float(q.x & 0xFFFF0000u) * x[j + 1]
         + __uint_as_float(q.y << 16)         * x[j + 2]
         + __uint_as_float(q.y & 0xFFFF0000u) * x[j + 3];
}

// y = Q x, rows striped over the 16 waves, 4 bf16 per lane per step (b64 loads).
__device__ __forceinline__ void matvec(const unsigned int* __restrict__ Qu,
                                       const float* __restrict__ x,
                                       float* __restrict__ yout, int tid)
{
    const int lane = tid & 31, wid = tid >> 5;
    for (int r = wid; r < NA; r += 16) {
        const unsigned int* rowu = Qu + (size_t)r * (NP / 2);
        const uint2* row = (const uint2*)rowu;
        if (r + 16 < NA)
            __builtin_prefetch((const void*)(rowu + 16 * (NP / 2)), 0, 1);
        float acc = 0.0f;
        const int jb = lane << 2;                 // 0..124, multiple of 4
#pragma unroll
        for (int s = 0; s < 3; ++s) {             // j <= 124+256 = 380 < 500: no guard
            const int j = jb + (s << 7);
            acc += dot4_bf16(row[j >> 2], x, j);
        }
        {
            const int j = jb + 384;               // up to 508: guard only here
            if (j < NA) acc += dot4_bf16(row[j >> 2], x, j);
        }
#pragma unroll
        for (int m = 16; m >= 1; m >>= 1) acc += __shfl_xor(acc, m, 32);
        if (lane == 0) yout[r] = acc;
    }
}

__global__ __launch_bounds__(512)
void qp_solver(const unsigned short* __restrict__ Q,
               const float* __restrict__ C,
               float* __restrict__ out)
{
    __shared__ float vecIn[NP];
    __shared__ float vecOut[NP];
    __shared__ float red[32];

    const int tid = threadIdx.x;
    const int b   = blockIdx.x;
    const unsigned int* Qu = (const unsigned int*)(Q + (size_t)b * NP * NP);
    const bool active = (tid < NA);
    const float creg = active ? C[(size_t)b * NA + tid] : 0.0f;

    // ---- power iteration for Lipschitz constant ----
    vecIn[tid] = active ? (1.0f / NA) : 0.0f;
    __syncthreads();
    for (int it = 0; it < 30; ++it) {
        matvec(Qu, vecIn, vecOut, tid);
        __syncthreads();
        const float u = active ? vecOut[tid] : 0.0f;
        const float n2 = blockSum(u * u, red, tid);
        vecIn[tid] = u / (sqrtf(n2) + 1e-12f);
        __syncthreads();
    }
    matvec(Qu, vecIn, vecOut, tid);
    __syncthreads();
    const float uq = active ? vecOut[tid] : 0.0f;
    const float L  = blockSum(vecIn[tid] * uq, red, tid);
    const float step = 1.0f / (L + 1e-8f);

    // ---- FISTA ----
    float w  = active ? (1.0f / NA) : 0.0f;
    float tk = 1.0f;
    __syncthreads();
    vecIn[tid] = w;                       // y0 = w0
    __syncthreads();

    for (int it = 0; it < 300; ++it) {
        matvec(Qu, vecIn, vecOut, tid);   // Qy
        __syncthreads();
        const float y = vecIn[tid];
        const float z = active ? (y - step * (vecOut[tid] - creg)) : -1e30f;

        // simplex projection: bisection for theta with sum(max(z-theta,0)) = 1
        const float zmax = blockMax(z, red, tid);
        float lo = zmax - 1.0f, hi = zmax;
#pragma unroll 1
        for (int bi = 0; bi < 26; ++bi) {
            const float th = 0.5f * (lo + hi);
            const float s  = blockSum(fmaxf(z - th, 0.0f), red, tid);
            if (s > 1.0f) lo = th; else hi = th;   // s is block-uniform
        }
        const float theta = 0.5f * (lo + hi);
        const float wn = fmaxf(z - theta, 0.0f);   // inactive lanes -> 0

        const float tn = 0.5f * (1.0f + sqrtf(1.0f + 4.0f * tk * tk));
        const float yn = wn + ((tk - 1.0f) / tn) * (wn - w);
        w = wn; tk = tn;
        vecIn[tid] = active ? yn : 0.0f;
        __syncthreads();
    }

    // ---- clamp + renormalize + store ----
    const float s = blockSum(w, red, tid);
    if (active) out[(size_t)b * NA + tid] = w / fmaxf(s, 1e-8f);
}

// ---------------------------------------------------------------------------
extern "C" void kernel_launch(void* const* d_in, const int* in_sizes, int n_in,
                              void* d_out, int out_size, void* d_ws, size_t ws_size,
                              hipStream_t stream)
{
    (void)in_sizes; (void)n_in; (void)out_size; (void)ws_size;
    const float* c_hat    = (const float*)d_in[0];   // (256, 500) f32
    const float* R_scaled = (const float*)d_in[1];   // (256, 252, 500) f32
    unsigned short* Q     = (unsigned short*)d_ws;   // (256, 512, 512) bf16

    // Phase 1: Q = R^T R per batch (WMMA bf16), 64x64 tiles.
    dim3 gQ(NP / 64, NP / 64, BATCH);   // (8, 8, 256)
    qp_build_Q<<<gQ, 256, 0, stream>>>(R_scaled, Q);

    // Phase 2: one persistent workgroup per batch element.
    qp_solver<<<BATCH, 512, 0, stream>>>(Q, c_hat, (float*)d_out);
}
